// OpenPanguMoE_9620726743828
// MI455X (gfx1250) — matrix operand order repaired and checked by reading, not compiled, not run on hardware
//
#include <hip/hip_runtime.h>
#include <hip/hip_bf16.h>

// ---------------- problem constants ----------------
#define HDIM   1024      // hidden size
#define IDIM   512       // routed intermediate
#define NEXP   8         // num routed experts
#define TOPK   2
#define SHI    1024      // shared intermediate (I * N_SHARED)
#define TTOK   2048      // B*S tokens
#define RSCALE 2.5f

// LDS row pitches padded by 8 bf16 (16B) so consecutive rows land on
// different bank groups (64 banks x 4B): pitch*2 % 256B == 16B -> 4-bank
// stagger per row -> conflict-free 16-lane b128 fragment loads.
#define XSP    (HDIM + 8)    // 1032
#define MIDPR  (IDIM + 8)    // 520
#define MIDPS  (SHI  + 8)    // 1032

typedef __attribute__((ext_vector_type(16))) __bf16 v16bf;
typedef __attribute__((ext_vector_type(8)))  __bf16 v8bf;
typedef __attribute__((ext_vector_type(4)))  __bf16 v4bf;
typedef __attribute__((ext_vector_type(8)))  float  v8f;

__device__ __forceinline__ float fast_sigmoid(float x) {
    return __builtin_amdgcn_rcpf(1.f + __expf(-x));   // v_rcp_f32, no divide chain
}

// ---------------- f32 -> bf16 convert (vectorized x4) ----------------
__global__ void pangu_f2bf4(const float4* __restrict__ src, v4bf* __restrict__ dst, int n4) {
    int i = blockIdx.x * 256 + threadIdx.x;
    if (i < n4) {
        float4 v = src[i];
        v4bf o;
        o[0] = (__bf16)v.x; o[1] = (__bf16)v.y; o[2] = (__bf16)v.z; o[3] = (__bf16)v.w;
        dst[i] = o;
    }
}

// ---------------- gating: sigmoid top-2, build expert lists ----------------
// one wave (32 lanes) per token; 8 tokens per 256-thread block
__global__ __launch_bounds__(256) void pangu_gate(
    const float* __restrict__ x,        // [T,H] f32
    const float* __restrict__ gw,       // [E,H] f32
    int*   __restrict__ cnt,            // [E]   (pre-zeroed)
    int*   __restrict__ list,           // [E,T] entries = t*2+k
    float* __restrict__ wts)            // [T*2] combine weights
{
    const int t    = blockIdx.x * 8 + (threadIdx.x >> 5);
    const int lane = threadIdx.x & 31;
    if (t >= TTOK) return;

    float acc[NEXP];
#pragma unroll
    for (int e = 0; e < NEXP; e++) acc[e] = 0.f;

    for (int h = lane; h < HDIM; h += 32) {
        float xv = x[(size_t)t * HDIM + h];
#pragma unroll
        for (int e = 0; e < NEXP; e++) acc[e] += xv * gw[e * HDIM + h];
    }
#pragma unroll
    for (int off = 16; off > 0; off >>= 1) {
#pragma unroll
        for (int e = 0; e < NEXP; e++) acc[e] += __shfl_xor(acc[e], off, 32);
    }
    if (lane == 0) {
        float sc[NEXP];
#pragma unroll
        for (int e = 0; e < NEXP; e++) sc[e] = fast_sigmoid(acc[e]);
        int i1 = 0;
#pragma unroll
        for (int e = 1; e < NEXP; e++) if (sc[e] > sc[i1]) i1 = e;
        int i2 = (i1 == 0) ? 1 : 0;
#pragma unroll
        for (int e = 0; e < NEXP; e++) if (e != i1 && sc[e] > sc[i2]) i2 = e;
        float w1 = sc[i1], w2 = sc[i2];
        float inv = RSCALE / (w1 + w2 + 1e-20f);
        w1 *= inv; w2 *= inv;

        int s1 = atomicAdd(&cnt[i1], 1);
        int s2 = atomicAdd(&cnt[i2], 1);
        list[i1 * TTOK + s1] = t * 2 + 0;
        list[i2 * TTOK + s2] = t * 2 + 1;
        wts[t * 2 + 0] = w1;
        wts[t * 2 + 1] = w2;
    }
}

// ---------------- helpers for WMMA fragments ----------------
__device__ __forceinline__ v16bf load_a_frag(const __bf16* rowbase, int k0) {
    // rowbase already offset by (lane&15)*pitch + (lane>>4)*16
    v8bf lo = *(const v8bf*)(rowbase + k0);
    v8bf hi = *(const v8bf*)(rowbase + k0 + 8);
    return __builtin_shufflevector(lo, hi, 0,1,2,3,4,5,6,7,8,9,10,11,12,13,14,15);
}

// ---------------- routed experts: gathered 16-token tiles ----------------
// grid = NEXP * 128; block = 256 (8 wave32)
__global__ __launch_bounds__(256) void pangu_routed(
    const __bf16* __restrict__ xbf,     // [T,H]
    const __bf16* __restrict__ wg,      // [E,I,H]
    const __bf16* __restrict__ wu,      // [E,I,H]
    const __bf16* __restrict__ wd,      // [E,H,I]
    const int*   __restrict__ cnt,      // [E]
    const int*   __restrict__ list,     // [E,T]
    const float* __restrict__ wts,      // [T*2]
    float*       __restrict__ routed)   // [T*2, H]
{
    const int e    = blockIdx.x >> 7;
    const int tile = blockIdx.x & 127;
    const int ntok = cnt[e];
    if (tile * 16 >= ntok) return;

    __shared__ __bf16 xs[16 * XSP];      // ~32.3 KB, bank-staggered rows
    __shared__ __bf16 mids[16 * MIDPR];  // ~16.3 KB
    __shared__ int   sEntry[16];
    __shared__ float sWt[16];

    const int tid = threadIdx.x;
    if (tid < 16) {
        int r = tile * 16 + tid;
        int entry = (r < ntok) ? list[e * TTOK + r] : -1;
        sEntry[tid] = entry;
        sWt[tid] = (entry >= 0) ? wts[entry] : 0.f;
    }
    __syncthreads();

    // gather 16 token rows into LDS as bf16 (16B chunks)
    for (int c = tid; c < 16 * (HDIM / 8); c += 256) {
        int row = c >> 7, chunk = c & 127;
        int entry = sEntry[row];
        uint4 v = make_uint4(0u, 0u, 0u, 0u);
        if (entry >= 0) {
            int t = entry >> 1;
            v = *(const uint4*)(xbf + (size_t)t * HDIM + chunk * 8);
        }
        *(uint4*)(&xs[row * XSP + chunk * 8]) = v;
    }
    __syncthreads();

    const int wave = tid >> 5;
    const int lane = tid & 31;
    const int lrow = lane & 15;
    const int half = lane >> 4;

    const __bf16* wge = wg + (size_t)e * IDIM * HDIM;
    const __bf16* wue = wu + (size_t)e * IDIM * HDIM;
    const __bf16* arowA = xs + lrow * XSP + half * 16;

    // Stage A: g = x@Wg^T, u = x@Wu^T, mid = silu(g)*u  (N = IDIM, A reused by 2 WMMAs)
    for (int nt = wave; nt < IDIM / 16; nt += 8) {
        const int n0 = nt * 16;
        v8f accg = {}, accu = {};
        const __bf16* bg = wge + (size_t)(n0 + lrow) * HDIM + half * 16;
        const __bf16* bu = wue + (size_t)(n0 + lrow) * HDIM + half * 16;
        __builtin_prefetch(wge + (size_t)(n0 + 128 + lrow) * HDIM + half * 16, 0, 3);
        __builtin_prefetch(wue + (size_t)(n0 + 128 + lrow) * HDIM + half * 16, 0, 3);
#pragma unroll 4
        for (int k0 = 0; k0 < HDIM; k0 += 32) {
            v16bf a  = load_a_frag(arowA, k0);
            v16bf fg = *(const v16bf*)(bg + k0);
            v16bf fu = *(const v16bf*)(bu + k0);
            accg = __builtin_amdgcn_wmma_f32_16x16x32_bf16(false, a, false, fg, (short)0, accg, false, false);
            accu = __builtin_amdgcn_wmma_f32_16x16x32_bf16(false, a, false, fu, (short)0, accu, false, false);
        }
#pragma unroll
        for (int r = 0; r < 8; r++) {
            float g = accg[r], u = accu[r];
            float m = g * fast_sigmoid(g) * u;
            mids[(r + half * 8) * MIDPR + n0 + lrow] = (__bf16)m;
        }
    }
    __syncthreads();

    // Stage B: y = mid @ Wd^T  (N = HDIM, K = IDIM); two n-tiles per pass -> A reused x2
    const __bf16* wde = wd + (size_t)e * HDIM * IDIM;
    const __bf16* arowB = mids + lrow * MIDPR + half * 16;
#pragma unroll
    for (int j = 0; j < 4; j++) {
        const int n0 = (wave + 16 * j) * 16;
        const int n1 = n0 + 128;                    // (wave + 8) tile
        v8f acc0 = {}, acc1 = {};
        const __bf16* b0 = wde + (size_t)(n0 + lrow) * IDIM + half * 16;
        const __bf16* b1 = wde + (size_t)(n1 + lrow) * IDIM + half * 16;
#pragma unroll 2
        for (int k0 = 0; k0 < IDIM; k0 += 32) {
            v16bf a  = load_a_frag(arowB, k0);
            v16bf f0 = *(const v16bf*)(b0 + k0);
            v16bf f1 = *(const v16bf*)(b1 + k0);
            acc0 = __builtin_amdgcn_wmma_f32_16x16x32_bf16(false, a, false, f0, (short)0, acc0, false, false);
            acc1 = __builtin_amdgcn_wmma_f32_16x16x32_bf16(false, a, false, f1, (short)0, acc1, false, false);
        }
#pragma unroll
        for (int r = 0; r < 8; r++) {
            int m = r + half * 8;
            int entry = sEntry[m];
            if (entry >= 0) {
                float w = sWt[m];
                routed[(size_t)entry * HDIM + n0 + lrow] = acc0[r] * w;
                routed[(size_t)entry * HDIM + n1 + lrow] = acc1[r] * w;
            }
        }
    }
}

// ---------------- shared expert: dense 16-token tiles -> d_out ----------------
// grid = T/16 = 128; block = 256
__global__ __launch_bounds__(256) void pangu_shared(
    const __bf16* __restrict__ xbf,     // [T,H]
    const __bf16* __restrict__ wg,      // [SHI,H]
    const __bf16* __restrict__ wu,      // [SHI,H]
    const __bf16* __restrict__ wd,      // [H,SHI]
    float*        __restrict__ out)     // [T,H]
{
    __shared__ __bf16 xs[16 * XSP];     // ~32.3 KB
    __shared__ __bf16 mids[16 * MIDPS]; // ~32.3 KB  (total ~64.5 KB; gfx1250 WG LDS <= 320 KB)

    const int tile = blockIdx.x;
    const int tid  = threadIdx.x;
    const int t0   = tile * 16;

    for (int c = tid; c < 16 * (HDIM / 8); c += 256) {
        int row = c >> 7, chunk = c & 127;
        *(uint4*)(&xs[row * XSP + chunk * 8]) =
            *(const uint4*)(xbf + (size_t)(t0 + row) * HDIM + chunk * 8);
    }
    __syncthreads();

    const int wave = tid >> 5;
    const int lane = tid & 31;
    const int lrow = lane & 15;
    const int half = lane >> 4;
    const __bf16* arowA = xs + lrow * XSP + half * 16;

    // Stage A: N = SHI = 1024 -> 64 n-tiles, A reused by gate+up WMMAs
    for (int nt = wave; nt < SHI / 16; nt += 8) {
        const int n0 = nt * 16;
        v8f accg = {}, accu = {};
        const __bf16* bg = wg + (size_t)(n0 + lrow) * HDIM + half * 16;
        const __bf16* bu = wu + (size_t)(n0 + lrow) * HDIM + half * 16;
        __builtin_prefetch(wg + (size_t)(n0 + 128 + lrow) * HDIM + half * 16, 0, 3);
        __builtin_prefetch(wu + (size_t)(n0 + 128 + lrow) * HDIM + half * 16, 0, 3);
#pragma unroll 4
        for (int k0 = 0; k0 < HDIM; k0 += 32) {
            v16bf a  = load_a_frag(arowA, k0);
            v16bf fg = *(const v16bf*)(bg + k0);
            v16bf fu = *(const v16bf*)(bu + k0);
            accg = __builtin_amdgcn_wmma_f32_16x16x32_bf16(false, a, false, fg, (short)0, accg, false, false);
            accu = __builtin_amdgcn_wmma_f32_16x16x32_bf16(false, a, false, fu, (short)0, accu, false, false);
        }
#pragma unroll
        for (int r = 0; r < 8; r++) {
            float g = accg[r], u = accu[r];
            float m = g * fast_sigmoid(g) * u;
            mids[(r + half * 8) * MIDPS + n0 + lrow] = (__bf16)m;
        }
    }
    __syncthreads();

    // Stage B: N = HDIM -> 64 n-tiles, K = SHI; two n-tiles per pass
    const __bf16* arowB = mids + lrow * MIDPS + half * 16;
#pragma unroll
    for (int j = 0; j < 4; j++) {
        const int n0 = (wave + 16 * j) * 16;
        const int n1 = n0 + 128;
        v8f acc0 = {}, acc1 = {};
        const __bf16* b0 = wd + (size_t)(n0 + lrow) * SHI + half * 16;
        const __bf16* b1 = wd + (size_t)(n1 + lrow) * SHI + half * 16;
#pragma unroll 2
        for (int k0 = 0; k0 < SHI; k0 += 32) {
            v16bf a  = load_a_frag(arowB, k0);
            v16bf f0 = *(const v16bf*)(b0 + k0);
            v16bf f1 = *(const v16bf*)(b1 + k0);
            acc0 = __builtin_amdgcn_wmma_f32_16x16x32_bf16(false, a, false, f0, (short)0, acc0, false, false);
            acc1 = __builtin_amdgcn_wmma_f32_16x16x32_bf16(false, a, false, f1, (short)0, acc1, false, false);
        }
#pragma unroll
        for (int r = 0; r < 8; r++) {
            int m = r + half * 8;
            out[(size_t)(t0 + m) * HDIM + n0 + lrow] = acc0[r];
            out[(size_t)(t0 + m) * HDIM + n1 + lrow] = acc1[r];
        }
    }
}

// ---------------- final combine: out += routed(t,0) + routed(t,1) ----------------
__global__ void pangu_combine(float* __restrict__ out, const float* __restrict__ routed) {
    int i = blockIdx.x * 256 + threadIdx.x;         // over T*H/4 float4s
    if (i >= TTOK * HDIM / 4) return;
    int t = i >> 8;                                  // 256 float4 per row
    int c = i & 255;
    const float4* r4 = (const float4*)routed;
    float4* o4 = (float4*)out;
    float4 o = o4[i];
    float4 a = r4[(size_t)(t * 2 + 0) * 256 + c];
    float4 b = r4[(size_t)(t * 2 + 1) * 256 + c];
    o.x += a.x + b.x; o.y += a.y + b.y; o.z += a.z + b.z; o.w += a.w + b.w;
    o4[i] = o;
}

// ---------------- host launch ----------------
static inline size_t align_up(size_t x, size_t a) { return (x + a - 1) / a * a; }

extern "C" void kernel_launch(void* const* d_in, const int* in_sizes, int n_in,
                              void* d_out, int out_size, void* d_ws, size_t ws_size,
                              hipStream_t stream) {
    const float* x_f   = (const float*)d_in[0];   // [T,H]
    const float* gw_f  = (const float*)d_in[1];   // [E,H]
    const float* wg_f  = (const float*)d_in[2];   // [E,I,H]
    const float* wu_f  = (const float*)d_in[3];   // [E,I,H]
    const float* wd_f  = (const float*)d_in[4];   // [E,H,I]
    const float* wsg_f = (const float*)d_in[5];   // [SHI,H]
    const float* wsu_f = (const float*)d_in[6];   // [SHI,H]
    const float* wsd_f = (const float*)d_in[7];   // [H,SHI]
    float* out = (float*)d_out;

    // workspace carve-up
    char* ws = (char*)d_ws;
    size_t off = 0;
    auto take = [&](size_t bytes) -> void* {
        void* p = ws + off; off = align_up(off + bytes, 256); return p;
    };
    __bf16* xbf  = (__bf16*)take((size_t)TTOK * HDIM * 2);
    __bf16* wgb  = (__bf16*)take((size_t)NEXP * IDIM * HDIM * 2);
    __bf16* wub  = (__bf16*)take((size_t)NEXP * IDIM * HDIM * 2);
    __bf16* wdb  = (__bf16*)take((size_t)NEXP * HDIM * IDIM * 2);
    __bf16* wsgb = (__bf16*)take((size_t)SHI * HDIM * 2);
    __bf16* wsub = (__bf16*)take((size_t)SHI * HDIM * 2);
    __bf16* wsdb = (__bf16*)take((size_t)HDIM * SHI * 2);
    float*  routed = (float*)take((size_t)TTOK * TOPK * HDIM * 4);
    int*    cnt    = (int*)take(NEXP * 4);
    int*    list   = (int*)take((size_t)NEXP * TTOK * 4);
    float*  wts    = (float*)take((size_t)TTOK * TOPK * 4);
    (void)ws_size; (void)n_in; (void)in_sizes; (void)out_size;

    auto conv = [&](const float* s, __bf16* d, int n) {
        int n4 = n / 4;
        pangu_f2bf4<<<(n4 + 255) / 256, 256, 0, stream>>>((const float4*)s, (v4bf*)d, n4);
    };
    conv(x_f,   xbf,  TTOK * HDIM);
    conv(wg_f,  wgb,  NEXP * IDIM * HDIM);
    conv(wu_f,  wub,  NEXP * IDIM * HDIM);
    conv(wd_f,  wdb,  NEXP * HDIM * IDIM);
    conv(wsg_f, wsgb, SHI * HDIM);
    conv(wsu_f, wsub, SHI * HDIM);
    conv(wsd_f, wsdb, HDIM * SHI);

    hipMemsetAsync(cnt, 0, NEXP * sizeof(int), stream);

    pangu_gate<<<TTOK / 8, 256, 0, stream>>>(x_f, gw_f, cnt, list, wts);

    pangu_shared<<<TTOK / 16, 256, 0, stream>>>(xbf, wsgb, wsub, wsdb, out);

    pangu_routed<<<NEXP * (TTOK / 16), 256, 0, stream>>>(xbf, wgb, wub, wdb,
                                                         cnt, list, wts, routed);

    pangu_combine<<<(TTOK * HDIM / 4 + 255) / 256, 256, 0, stream>>>(out, routed);
}
